// AsyNonLocal2D_11115375362195
// MI455X (gfx1250) — compile-verified, matrix operand order checked by reading, not run on hardware
//
#include <hip/hip_runtime.h>

// ---------------------------------------------------------------------------
// CDNA5 (gfx1250) fused non-local attention.
// Precision: bf16 operands, f32 accumulation via v_wmma_f32_16x16x32_bf16.
// Register-fat kernels: __launch_bounds__(128,1) to prevent scratch spills.
// ---------------------------------------------------------------------------

typedef __attribute__((ext_vector_type(16))) __bf16 v16bf;
typedef __attribute__((ext_vector_type(8)))  __bf16 v8bf;
typedef __attribute__((ext_vector_type(8)))  float  v8f;

static __device__ __forceinline__ __bf16 f2bf(float f) {
  union { float f; unsigned u; } v; v.f = f;
  unsigned r = v.u + 0x7fffu + ((v.u >> 16) & 1u);   // round-to-nearest-even
  unsigned short h = (unsigned short)(r >> 16);
  return __builtin_bit_cast(__bf16, h);
}

// A/B fragment from a row-major bf16 matrix [16 x >=32], leading dim ld (elems).
// Per ISA 7.12.2 (16-bit A 16x32): lane L holds row L&15; elements 0..7 are
// K = kb..kb+7, elements 8..15 are K = kb+16..kb+23, kb = (L>=16)*8.
static __device__ __forceinline__ v16bf frag_bf16_rm(const __bf16* __restrict__ base, int ld) {
  const int lane = threadIdx.x & 31;
  const int r  = lane & 15;
  const int kb = (lane >> 4) << 3;
  const __bf16* p = base + (size_t)r * ld + kb;
  v8bf lo = *(const v8bf*)(p);
  v8bf hi = *(const v8bf*)(p + 16);
  return __builtin_shufflevector(lo, hi, 0,1,2,3,4,5,6,7,8,9,10,11,12,13,14,15);
}

// Same fragment, but source is row-major f32 (weights) -> convert to bf16.
static __device__ __forceinline__ v16bf frag_f32_rm(const float* __restrict__ base, int ld) {
  const int lane = threadIdx.x & 31;
  const int r  = lane & 15;
  const int kb = (lane >> 4) << 3;
  const float* p = base + (size_t)r * ld + kb;
  v16bf f;
#pragma unroll
  for (int e = 0; e < 8; ++e) f[e] = f2bf(p[e]);
#pragma unroll
  for (int e = 0; e < 8; ++e) f[8 + e] = f2bf(p[16 + e]);
  return f;
}

// Fragment whose 16-wide dimension is contiguous but K is strided by ld
// (i.e. a tile of the *transpose* of a row-major [K x N] f32 matrix).
static __device__ __forceinline__ v16bf frag_f32_strided(const float* __restrict__ base, int ld) {
  const int lane = threadIdx.x & 31;
  const int r  = lane & 15;
  const int kb = (lane >> 4) << 3;
  const float* p = base + (size_t)kb * ld + r;
  v16bf f;
#pragma unroll
  for (int e = 0; e < 8; ++e) f[e] = f2bf(p[(size_t)e * ld]);
#pragma unroll
  for (int e = 0; e < 8; ++e) f[8 + e] = f2bf(p[(size_t)(16 + e) * ld]);
  return f;
}

static __device__ __forceinline__ v8f wmma_bf16(v16bf a, v16bf b, v8f c) {
  return __builtin_amdgcn_wmma_f32_16x16x32_bf16(false, a, false, b, (short)0, c, false, false);
}

// ---------------------------------------------------------------------------
// Projection -> position-major bf16 output [B, HW, 128] (theta, phi).
// out[q, c'] = sum_c act[c, q] * W[c', c] + bias[c'].
// ---------------------------------------------------------------------------
template <int NK>
__global__ __launch_bounds__(128, 1)
void proj_pos_kernel(const float* __restrict__ act, const float* __restrict__ W,
                     const float* __restrict__ bias, __bf16* __restrict__ out,
                     int HW) {
  const int lane = threadIdx.x & 31;
  const int gw   = blockIdx.x * 4 + (threadIdx.x >> 5);
  const int qpb  = HW >> 4;
  const int b    = gw / qpb;
  const int q0   = (gw % qpb) << 4;
  const int Cin  = NK * 32;

  v16bf a[NK];
#pragma unroll
  for (int kk = 0; kk < NK; ++kk)
    a[kk] = frag_f32_strided(act + (size_t)(b * Cin + kk * 32) * HW + q0, HW);

  const int coln  = lane & 15;
  const int rbase = (lane >> 4) << 3;
#pragma unroll
  for (int ct = 0; ct < 8; ++ct) {
    v8f acc = {};
#pragma unroll
    for (int kk = 0; kk < NK; ++kk) {
      v16bf bf = frag_f32_rm(W + (size_t)(ct * 16) * Cin + kk * 32, Cin);
      acc = wmma_bf16(a[kk], bf, acc);
    }
    const float bv = bias[ct * 16 + coln];
#pragma unroll
    for (int v = 0; v < 8; ++v) {
      const int row = rbase + v;
      out[((size_t)b * HW + q0 + row) * 128 + ct * 16 + coln] = f2bf(acc[v] + bv);
    }
  }
}

// ---------------------------------------------------------------------------
// Projection -> channel-major bf16 output [B, 128, HW] (g).
// out[c', q] = sum_c W[c', c] * act[c, q] + bias[c'].
// ---------------------------------------------------------------------------
template <int NK>
__global__ __launch_bounds__(128, 1)
void proj_chan_kernel(const float* __restrict__ act, const float* __restrict__ W,
                      const float* __restrict__ bias, __bf16* __restrict__ out,
                      int HW) {
  const int lane = threadIdx.x & 31;
  const int gw   = blockIdx.x * 4 + (threadIdx.x >> 5);
  const int qpb  = HW >> 4;
  const int b    = gw / qpb;
  const int q0   = (gw % qpb) << 4;
  const int Cin  = NK * 32;

  v8f acc[8];
#pragma unroll
  for (int i = 0; i < 8; ++i) acc[i] = (v8f){};

#pragma unroll
  for (int kk = 0; kk < NK; ++kk) {
    v16bf bfr = frag_f32_strided(act + (size_t)(b * Cin + kk * 32) * HW + q0, HW);
#pragma unroll
    for (int ct = 0; ct < 8; ++ct) {
      v16bf afr = frag_f32_rm(W + (size_t)(ct * 16) * Cin + kk * 32, Cin);
      acc[ct] = wmma_bf16(afr, bfr, acc[ct]);
    }
  }

  const int coln  = lane & 15;
  const int rbase = (lane >> 4) << 3;
#pragma unroll
  for (int ct = 0; ct < 8; ++ct) {
#pragma unroll
    for (int v = 0; v < 8; ++v) {
      const int row = ct * 16 + rbase + v;
      out[((size_t)b * 128 + row) * HW + q0 + coln] = f2bf(acc[ct][v] + bias[row]);
    }
  }
}

// ---------------------------------------------------------------------------
// Flash attention: per wave, one 16-row q strip; stream r in steps of 32.
// thetaT/phiT: [B, HW, 128] bf16 (position-major); g: [B, 128, HW] bf16.
// Software-pipelined: all 16 B-fragments (phi + g) for a step are issued up
// front, so g loads overlap the entire softmax section; softmax denominator
// is accumulated with WMMA (P @ ones) — only row-max uses lane shuffles.
// ---------------------------------------------------------------------------
__global__ __launch_bounds__(128, 1)
void attn_kernel(const __bf16* __restrict__ thetaT, const __bf16* __restrict__ phiT,
                 const __bf16* __restrict__ gmat, __bf16* __restrict__ y, int HW) {
  __shared__ __align__(16) __bf16 ptile[4][16 * 32];

  const int lane = threadIdx.x & 31;
  const int w    = threadIdx.x >> 5;
  const int gw   = blockIdx.x * 4 + w;
  const int qpb  = HW >> 4;
  const int b    = gw / qpb;
  const int q0   = (gw % qpb) << 4;
  const float sc = 0.08838834764831845f;  // 1/sqrt(128)

  // Constant ones fragment (row-sum via WMMA).
  v16bf ones;
#pragma unroll
  for (int e = 0; e < 16; ++e) ones[e] = f2bf(1.0f);

  // Resident theta fragments: 16 q rows x 128 channels.
  v16bf ath[4];
#pragma unroll
  for (int kk = 0; kk < 4; ++kk)
    ath[kk] = frag_bf16_rm(thetaT + ((size_t)b * HW + q0) * 128 + kk * 32, 128);

  v8f o[8];
  v8f lacc = (v8f){};
  float m[8];
#pragma unroll
  for (int i = 0; i < 8; ++i) { o[i] = (v8f){}; m[i] = -3.0e38f; }

  const int coln  = lane & 15;
  const int rbase = (lane >> 4) << 3;
  __bf16* myp = &ptile[w][0];

  for (int r0 = 0; r0 < HW; r0 += 32) {
    if (r0 + 32 < HW) {
      __builtin_prefetch(phiT + ((size_t)b * HW + r0 + 32) * 128, 0, 3);   // global_prefetch_b8
      __builtin_prefetch(gmat + ((size_t)b * 128) * HW + r0 + 32, 0, 3);
    }

    // ---- Issue ALL B-fragment loads for this step up front. ----
    v16bf pb[8];   // phi fragments: [0..3] cols r0..r0+15, [4..7] cols +16
    v16bf gb[8];   // g fragments for the 8 c'-tiles
#pragma unroll
    for (int kk = 0; kk < 4; ++kk) {
      pb[kk]     = frag_bf16_rm(phiT + ((size_t)b * HW + r0) * 128 + kk * 32, 128);
      pb[4 + kk] = frag_bf16_rm(phiT + ((size_t)b * HW + r0 + 16) * 128 + kk * 32, 128);
    }
#pragma unroll
    for (int ct = 0; ct < 8; ++ct)
      gb[ct] = frag_bf16_rm(gmat + ((size_t)b * 128 + ct * 16) * HW + r0, HW);

    // ---- S = theta @ phi^T (16 x 32 score tile). ----
    v8f s0 = (v8f){}, s1 = (v8f){};
#pragma unroll
    for (int kk = 0; kk < 4; ++kk) {
      s0 = wmma_bf16(ath[kk], pb[kk],     s0);
      s1 = wmma_bf16(ath[kk], pb[4 + kk], s1);
    }

    // ---- Row max over the 32 new columns (batched butterfly). ----
    float a0[8], a1[8], mx[8];
#pragma unroll
    for (int v = 0; v < 8; ++v) {
      a0[v] = s0[v] * sc;
      a1[v] = s1[v] * sc;
      mx[v] = fmaxf(a0[v], a1[v]);
    }
#pragma unroll
    for (int mask = 1; mask <= 8; mask <<= 1) {
      float t[8];
#pragma unroll
      for (int v = 0; v < 8; ++v) t[v] = __shfl_xor(mx[v], mask, 32);
#pragma unroll
      for (int v = 0; v < 8; ++v) mx[v] = fmaxf(mx[v], t[v]);
    }

    // ---- Online softmax; write P (bf16) into per-wave LDS 16x32 tile. ----
    float corr[8];
#pragma unroll
    for (int v = 0; v < 8; ++v) {
      const float nm = fmaxf(m[v], mx[v]);
      corr[v] = __expf(m[v] - nm);
      m[v] = nm;
      const float p0 = __expf(a0[v] - nm);
      const float p1 = __expf(a1[v] - nm);
      const int row = rbase + v;
      myp[row * 32 + coln]      = f2bf(p0);
      myp[row * 32 + 16 + coln] = f2bf(p1);
    }

    // Rescale running O and the denominator accumulator (g loads in flight).
#pragma unroll
    for (int ct = 0; ct < 8; ++ct)
#pragma unroll
      for (int v = 0; v < 8; ++v) o[ct][v] *= corr[v];
#pragma unroll
    for (int v = 0; v < 8; ++v) lacc[v] *= corr[v];

    // P back out of LDS in A-fragment layout (C->A transpose bounce).
    v16bf ap = frag_bf16_rm(myp, 32);

    // Denominator: l += P @ ones  (one WMMA replaces 32 lane shuffles).
    lacc = wmma_bf16(ap, ones, lacc);

    // O += P @ g.
#pragma unroll
    for (int ct = 0; ct < 8; ++ct)
      o[ct] = wmma_bf16(ap, gb[ct], o[ct]);
  }

  float linv[8];
#pragma unroll
  for (int v = 0; v < 8; ++v) linv[v] = 1.f / lacc[v];
#pragma unroll
  for (int ct = 0; ct < 8; ++ct)
#pragma unroll
    for (int v = 0; v < 8; ++v) {
      const int row = rbase + v;
      y[((size_t)b * HW + q0 + row) * 128 + ct * 16 + coln] = f2bf(o[ct][v] * linv[v]);
    }
}

// ---------------------------------------------------------------------------
// out[co, q] = query[co, q] + bo[co] + sum_c' Wo[co, c'] * y[q, c'].
// ---------------------------------------------------------------------------
__global__ __launch_bounds__(128, 1)
void outproj_kernel(const __bf16* __restrict__ y, const float* __restrict__ Wo,
                    const float* __restrict__ bo, const float* __restrict__ query,
                    float* __restrict__ out, int HW) {
  const int lane = threadIdx.x & 31;
  const int gw   = blockIdx.x * 4 + (threadIdx.x >> 5);
  const int qpb  = HW >> 4;
  const int b    = gw / qpb;
  const int q0   = (gw % qpb) << 4;

  v8f acc[16];
#pragma unroll
  for (int i = 0; i < 16; ++i) acc[i] = (v8f){};

#pragma unroll
  for (int kk = 0; kk < 4; ++kk) {
    v16bf bf = frag_bf16_rm(y + ((size_t)b * HW + q0) * 128 + kk * 32, 128);
#pragma unroll
    for (int mt = 0; mt < 16; ++mt) {
      v16bf af = frag_f32_rm(Wo + (size_t)(mt * 16) * 128 + kk * 32, 128);
      acc[mt] = wmma_bf16(af, bf, acc[mt]);
    }
  }

  const int coln  = lane & 15;
  const int rbase = (lane >> 4) << 3;
#pragma unroll
  for (int mt = 0; mt < 16; ++mt) {
#pragma unroll
    for (int v = 0; v < 8; ++v) {
      const int co = mt * 16 + rbase + v;
      const size_t idx = ((size_t)b * 256 + co) * HW + q0 + coln;
      out[idx] = acc[mt][v] + query[idx] + bo[co];
    }
  }
}

// ---------------------------------------------------------------------------
extern "C" void kernel_launch(void* const* d_in, const int* in_sizes, int n_in,
                              void* d_out, int out_size, void* d_ws, size_t ws_size,
                              hipStream_t stream) {
  (void)in_sizes; (void)n_in; (void)out_size; (void)ws_size;
  const float* query = (const float*)d_in[0];   // [4,256,64,64]
  const float* refr  = (const float*)d_in[1];   // [4,512,64,64]
  const float* Wg    = (const float*)d_in[2];   // [128,512]
  const float* bg    = (const float*)d_in[3];
  const float* Wt    = (const float*)d_in[4];   // [128,256]
  const float* bt    = (const float*)d_in[5];
  const float* Wp    = (const float*)d_in[6];   // [128,512]
  const float* bp    = (const float*)d_in[7];
  const float* Wo    = (const float*)d_in[8];   // [256,128]
  const float* bo    = (const float*)d_in[9];
  float* out = (float*)d_out;

  const int B = 4, HW = 4096, CI = 128;
  const size_t tensElems = (size_t)B * HW * CI;       // 2M elems, 4 MB bf16 each
  __bf16* thetaT = reinterpret_cast<__bf16*>(d_ws);
  __bf16* phiT   = thetaT + tensElems;
  __bf16* gbuf   = phiT + tensElems;
  __bf16* ybuf   = gbuf + tensElems;

  const dim3 blk(128);                                 // 4 waves (wave32)
  const dim3 grid((B * HW / 16) / 4);                  // 1024 wave-strips -> 256 blocks

  proj_pos_kernel<8>  <<<grid, blk, 0, stream>>>(query, Wt, bt, thetaT, HW);  // theta (Cin=256)
  proj_pos_kernel<16> <<<grid, blk, 0, stream>>>(refr,  Wp, bp, phiT,  HW);   // phi   (Cin=512)
  proj_chan_kernel<16><<<grid, blk, 0, stream>>>(refr,  Wg, bg, gbuf,  HW);   // g     (Cin=512)
  attn_kernel          <<<grid, blk, 0, stream>>>(thetaT, phiT, gbuf, ybuf, HW);
  outproj_kernel       <<<grid, blk, 0, stream>>>(ybuf, Wo, bo, query, out, HW);
}
